// CachedAttention_2052994367817
// MI455X (gfx1250) — compile-verified, matrix operand order checked by reading, not run on hardware
//
#include <hip/hip_runtime.h>
#include <hip/hip_bf16.h>

// ---------------------------------------------------------------------------
// CDNA5 (gfx1250) cached-attention: bf16 WMMA pipeline, f32 accumulate.
// B=4, QLEN=2048, PAST=2048, D=2048, T=PAST+QLEN=4096.
// GEMM staging now uses gfx1250 async global->LDS (ASYNCcnt) + double buffer.
// ---------------------------------------------------------------------------

typedef __attribute__((ext_vector_type(16))) __bf16 v16bf;
typedef __attribute__((ext_vector_type(8)))  float  v8f;

static constexpr int    kB    = 4;
static constexpr int    kQ    = 2048;
static constexpr int    kPAST = 2048;
static constexpr int    kD    = 2048;
static constexpr int    kT    = 4096;   // PAST + QLEN
static constexpr int    kM    = kB * kQ; // 8192 folded query rows

__device__ __forceinline__ unsigned short f2bf(float f) {
    union { float f; unsigned u; } x; x.f = f;
    unsigned r = x.u + 0x7FFFu + ((x.u >> 16) & 1u);  // RNE
    return (unsigned short)(r >> 16);
}

// ---------------------------------------------------------------------------
// Cast kernels (f32 -> bf16 bits), with layout remaps for the KV cache.
// ---------------------------------------------------------------------------
__global__ __launch_bounds__(256)
void cast_plain(const float* __restrict__ s, unsigned short* __restrict__ d, size_t n) {
    size_t i = (size_t)blockIdx.x * 256 + threadIdx.x;
    size_t stride = (size_t)gridDim.x * 256;
    for (; i < n; i += stride) d[i] = f2bf(s[i]);
}

// past_k [B][PAST][D] -> Kfull [B][T][D] rows [0, PAST)
__global__ __launch_bounds__(256)
void cast_pastk(const float* __restrict__ s, unsigned short* __restrict__ d) {
    size_t n = (size_t)kB * kPAST * kD;
    size_t per = (size_t)kPAST * kD;
    size_t i = (size_t)blockIdx.x * 256 + threadIdx.x;
    size_t stride = (size_t)gridDim.x * 256;
    for (; i < n; i += stride) {
        size_t b = i / per, rem = i - b * per;
        d[b * (size_t)kT * kD + rem] = f2bf(s[i]);
    }
}

// past_v [B][PAST][D] -> Vt [B][D][T] cols [0, PAST)
__global__ __launch_bounds__(256)
void cast_pastv_t(const float* __restrict__ s, unsigned short* __restrict__ d) {
    size_t n = (size_t)kB * kPAST * kD;
    size_t per = (size_t)kPAST * kD;
    size_t i = (size_t)blockIdx.x * 256 + threadIdx.x;
    size_t stride = (size_t)gridDim.x * 256;
    for (; i < n; i += stride) {
        size_t b = i / per, rem = i - b * per;
        size_t t = rem / kD, dd = rem - t * kD;
        d[b * (size_t)kD * kT + dd * (size_t)kT + t] = f2bf(s[i]);
    }
}

// ---------------------------------------------------------------------------
// NT GEMM: C[M,N] = scale * A[M,K] @ B[N,K]^T, bf16 in, f32 accumulate.
// Block tile 128x128, BK=32. 8 waves, each wave 32x64 = 2x4 WMMA 16x16 tiles.
// Staging: global_load_async_to_lds_b128 (per-lane), double-buffered LDS,
// one s_wait_asynccnt + one barrier per k-step.
// CT = unsigned short (bf16 out) or float. TRANSC writes C[N,M] instead.
// ---------------------------------------------------------------------------
static constexpr int BM = 128, BN = 128, BK = 32;
static constexpr int LSTR = 48;  // padded LDS row stride (ushorts) = 96 B: 16B-aligned rows

union Frag { v16bf v; unsigned u[8]; };

template <typename CT, bool TRANSC>
__global__ __launch_bounds__(256)
void gemm_nt(const unsigned short* __restrict__ A, int lda,
             const unsigned short* __restrict__ B, int ldb,
             CT* __restrict__ C, int ldc,
             int K, float scale)
{
    __shared__ __align__(16) unsigned short lA[2][BM * LSTR];
    __shared__ __align__(16) unsigned short lB[2][BN * LSTR];

    const int tid  = threadIdx.x;
    const int lane = tid & 31;
    const int wave = tid >> 5;
    const int wm   = (wave >> 1) * 32;   // wave row offset in block tile
    const int wn   = (wave & 1)  * 64;   // wave col offset in block tile
    const size_t rowBase = (size_t)blockIdx.y * BM;
    const size_t colBase = (size_t)blockIdx.x * BN;

    // Issue async global->LDS B128 loads for one 128x32 A tile + B tile.
    // 512 16-byte chunks per matrix / 256 threads = 2 per thread per matrix.
    auto issue_tiles = [&](int bufIdx, int k0) {
        #pragma unroll
        for (int i = 0; i < 2; ++i) {
            int c   = tid + i * 256;
            int row = c >> 2;             // 4 chunks per 32-wide row
            int col = (c & 3) << 3;       // 8 bf16 per chunk
            unsigned la = (unsigned)(size_t)&lA[bufIdx][row * LSTR + col];
            unsigned long long ga =
                (unsigned long long)(size_t)&A[(rowBase + row) * (size_t)lda + k0 + col];
            asm volatile("global_load_async_to_lds_b128 %0, %1, off"
                         :: "v"(la), "v"(ga) : "memory");
            unsigned lb = (unsigned)(size_t)&lB[bufIdx][row * LSTR + col];
            unsigned long long gb =
                (unsigned long long)(size_t)&B[(colBase + row) * (size_t)ldb + k0 + col];
            asm volatile("global_load_async_to_lds_b128 %0, %1, off"
                         :: "v"(lb), "v"(gb) : "memory");
        }
    };

    v8f acc[2][4];
    const v8f vzero = {0.f, 0.f, 0.f, 0.f, 0.f, 0.f, 0.f, 0.f};
    for (int i = 0; i < 2; ++i)
        for (int j = 0; j < 4; ++j) acc[i][j] = vzero;

    const int r16 = lane & 15;            // M (or N) index within 16
    const int kb  = (lane >> 4) << 3;     // K base: 0 or 8 per ISA A/B layout

    issue_tiles(0, 0);                    // prologue
    int cur = 0;

    for (int k0 = 0; k0 < K; k0 += BK) {
        // All of this wave's async chunks landed; barrier covers other waves'.
        asm volatile("s_wait_asynccnt 0x0" ::: "memory");
        __syncthreads();

        if (k0 + BK < K) issue_tiles(cur ^ 1, k0 + BK);   // overlap next tile
        if (k0 + 2 * BK < K) {                            // L2 warming 2 ahead
            __builtin_prefetch(&A[(rowBase + (tid >> 1)) * (size_t)lda + k0 + 2 * BK], 0, 1);
            __builtin_prefetch(&B[(colBase + (tid >> 1)) * (size_t)ldb + k0 + 2 * BK], 0, 1);
        }

        // Build fragments per the 16-bit 16x32 A / 32x16 B VGPR layouts:
        // elements 0..7 -> K = kb..kb+7, elements 8..15 -> K = kb+16..kb+23.
        Frag a[2], b[4];
        #pragma unroll
        for (int mi = 0; mi < 2; ++mi) {
            const unsigned short* p = &lA[cur][(wm + mi * 16 + r16) * LSTR];
            #pragma unroll
            for (int j = 0; j < 4; ++j) {
                a[mi].u[j]     = *(const unsigned*)&p[kb + 2 * j];
                a[mi].u[4 + j] = *(const unsigned*)&p[kb + 16 + 2 * j];
            }
        }
        #pragma unroll
        for (int ni = 0; ni < 4; ++ni) {
            const unsigned short* p = &lB[cur][(wn + ni * 16 + r16) * LSTR];
            #pragma unroll
            for (int j = 0; j < 4; ++j) {
                b[ni].u[j]     = *(const unsigned*)&p[kb + 2 * j];
                b[ni].u[4 + j] = *(const unsigned*)&p[kb + 16 + 2 * j];
            }
        }
        #pragma unroll
        for (int mi = 0; mi < 2; ++mi)
            #pragma unroll
            for (int ni = 0; ni < 4; ++ni)
                acc[mi][ni] = __builtin_amdgcn_wmma_f32_16x16x32_bf16(
                    false, a[mi].v, false, b[ni].v,
                    (short)0, acc[mi][ni], false, false);

        cur ^= 1;
    }

    // C layout: VGPR v, lanes 0-15 -> M=v, lanes 16-31 -> M=8+v; N = lane%16.
    const int rb = (lane >> 4) << 3;
    #pragma unroll
    for (int mi = 0; mi < 2; ++mi)
        #pragma unroll
        for (int ni = 0; ni < 4; ++ni)
            #pragma unroll
            for (int v = 0; v < 8; ++v) {
                size_t row = rowBase + wm + mi * 16 + rb + v;
                size_t col = colBase + wn + ni * 16 + (lane & 15);
                float val = acc[mi][ni][v] * scale;
                size_t idx = TRANSC ? col * (size_t)ldc + row
                                    : row * (size_t)ldc + col;
                if constexpr (sizeof(CT) == 2) ((unsigned short*)C)[idx] = f2bf(val);
                else                           ((float*)C)[idx] = val;
            }
}

// ---------------------------------------------------------------------------
// Row softmax over 4096 f32 scores -> bf16 probabilities. One block per row.
// ---------------------------------------------------------------------------
__global__ __launch_bounds__(256)
void softmax_rows(const float* __restrict__ S, unsigned short* __restrict__ P) {
    const size_t row = blockIdx.x;
    const float* r = S + row * (size_t)kT;
    unsigned short* o = P + row * (size_t)kT;
    const int tid = threadIdx.x, lane = tid & 31, wave = tid >> 5;

    float e[kT / 256];
    float m = -3.4e38f;
    #pragma unroll
    for (int j = 0; j < kT / 256; ++j) { e[j] = r[tid + j * 256]; m = fmaxf(m, e[j]); }
    #pragma unroll
    for (int off = 16; off; off >>= 1) m = fmaxf(m, __shfl_xor(m, off, 32));

    __shared__ float red[8];
    if (lane == 0) red[wave] = m;
    __syncthreads();
    float gm = red[0];
    #pragma unroll
    for (int i = 1; i < 8; ++i) gm = fmaxf(gm, red[i]);
    __syncthreads();

    float s = 0.f;
    #pragma unroll
    for (int j = 0; j < kT / 256; ++j) { e[j] = __expf(e[j] - gm); s += e[j]; }
    #pragma unroll
    for (int off = 16; off; off >>= 1) s += __shfl_xor(s, off, 32);
    if (lane == 0) red[wave] = s;
    __syncthreads();
    float tot = 0.f;
    #pragma unroll
    for (int i = 0; i < 8; ++i) tot += red[i];
    float inv = 1.f / tot;
    #pragma unroll
    for (int j = 0; j < kT / 256; ++j) o[tid + j * 256] = f2bf(e[j] * inv);
}

// ---------------------------------------------------------------------------
// Launch
// ---------------------------------------------------------------------------
extern "C" void kernel_launch(void* const* d_in, const int* in_sizes, int n_in,
                              void* d_out, int out_size, void* d_ws, size_t ws_size,
                              hipStream_t stream) {
    const float* x      = (const float*)d_in[0];
    const float* past_k = (const float*)d_in[1];
    const float* past_v = (const float*)d_in[2];
    const float* Wq     = (const float*)d_in[3];
    const float* Wk     = (const float*)d_in[4];
    const float* Wv     = (const float*)d_in[5];
    float* out = (float*)d_out;

    // Workspace carve-up (bf16 stages + f32 scores).
    char* w = (char*)d_ws;
    unsigned short* Xb  = (unsigned short*)w; w += (size_t)kM * kD * 2;        // 32 MB
    unsigned short* Wqb = (unsigned short*)w; w += (size_t)kD * kD * 2;        //  8 MB
    unsigned short* Wkb = (unsigned short*)w; w += (size_t)kD * kD * 2;        //  8 MB
    unsigned short* Wvb = (unsigned short*)w; w += (size_t)kD * kD * 2;        //  8 MB
    unsigned short* Qb  = (unsigned short*)w; w += (size_t)kM * kD * 2;        // 32 MB
    unsigned short* Kf  = (unsigned short*)w; w += (size_t)kB * kT * kD * 2;   // 64 MB
    unsigned short* Vt  = (unsigned short*)w; w += (size_t)kB * kD * kT * 2;   // 64 MB
    float*          Sf  = (float*)w;          w += (size_t)kB * kQ * kT * 4;   // 128 MB
    unsigned short* Pb  = (unsigned short*)w;                                  // 64 MB

    const float inv_sqrt_d = 0.022097086912079608f;  // 1/sqrt(2048)

    // 1) Downconvert operands.
    cast_plain<<<4096, 256, 0, stream>>>(x,  Xb,  (size_t)kM * kD);
    cast_plain<<<2048, 256, 0, stream>>>(Wq, Wqb, (size_t)kD * kD);
    cast_plain<<<2048, 256, 0, stream>>>(Wk, Wkb, (size_t)kD * kD);
    cast_plain<<<2048, 256, 0, stream>>>(Wv, Wvb, (size_t)kD * kD);
    cast_pastk  <<<4096, 256, 0, stream>>>(past_k, Kf);
    cast_pastv_t<<<4096, 256, 0, stream>>>(past_v, Vt);

    // 2) Q = X @ Wq^T  (bf16 out), all batches folded: M=8192.
    gemm_nt<unsigned short, false><<<dim3(kD / BN, kM / BM), 256, 0, stream>>>(
        Xb, kD, Wqb, kD, Qb, kD, kD, 1.f);

    // 3) K/V projections per batch into cache layouts.
    for (int b = 0; b < kB; ++b) {
        const unsigned short* Ab = Xb + (size_t)b * kQ * kD;
        // K_new rows -> Kfull[b][PAST..T)
        gemm_nt<unsigned short, false><<<dim3(kD / BN, kQ / BM), 256, 0, stream>>>(
            Ab, kD, Wkb, kD,
            Kf + (size_t)b * kT * kD + (size_t)kPAST * kD, kD, kD, 1.f);
        // V_new transposed -> Vt[b][:, PAST..T)
        gemm_nt<unsigned short, true><<<dim3(kD / BN, kQ / BM), 256, 0, stream>>>(
            Ab, kD, Wvb, kD,
            Vt + (size_t)b * kD * kT + kPAST, kT, kD, 1.f);
    }

    // 4) Scores = (Q @ Kfull^T) / sqrt(D), f32 out.
    for (int b = 0; b < kB; ++b)
        gemm_nt<float, false><<<dim3(kT / BN, kQ / BM), 256, 0, stream>>>(
            Qb + (size_t)b * kQ * kD, kD,
            Kf + (size_t)b * kT * kD, kD,
            Sf + (size_t)b * kQ * kT, kT, kD, inv_sqrt_d);

    // 5) Row softmax -> bf16 probabilities.
    softmax_rows<<<kB * kQ, 256, 0, stream>>>(Sf, Pb);

    // 6) Output = P @ Vt^T  (f32 out).
    for (int b = 0; b < kB; ++b)
        gemm_nt<float, false><<<dim3(kD / BN, kQ / BM), 256, 0, stream>>>(
            Pb + (size_t)b * kQ * kT, kT,
            Vt + (size_t)b * kD * kT, kT,
            out + (size_t)b * kQ * kD, kD, kT, 1.f);
}
